// LSTM_88553635709690
// MI455X (gfx1250) — compile-verified, matrix operand order checked by reading, not run on hardware
//
#include <hip/hip_runtime.h>
#include <math.h>

typedef __attribute__((ext_vector_type(16))) _Float16 v16h;
typedef __attribute__((ext_vector_type(8)))  _Float16 v8h;
typedef __attribute__((ext_vector_type(8)))  float    v8f;

#define Bsz 256
#define Tn  1024
#define Hn  256

// workspace layout (byte offsets)
#define WCAT0_OFF 0u          // f16 [1024][288] : cols 0..255 = W_hh0, 256..271 = W_ih0, 272..287 = 0
#define WCAT1_OFF 589824u     // f16 [1024][512] : cols 0..255 = W_hh1, 256..511 = W_ih1
#define FC1W_OFF  1638400u    // f16 [256][256]
#define BIAS0_OFF 1769472u    // f32 [1024] = b_ih0 + b_hh0
#define BIAS1_OFF 1773568u    // f32 [1024]
#define FC2W_OFF  1777664u    // f32 [256]
#define XPACK_OFF 4194304u    // f16 [T][B][32] (cols 16..31 zero)

#define P0 296   // LDS pitch (halves) for layer0 A tile [16][288]+pad
#define P1 520   // LDS pitch (halves) for layer1 A tile [16][512]+pad

static __device__ __forceinline__ v16h mk16(v8h lo, v8h hi) {
  v16h r;
#pragma unroll
  for (int i = 0; i < 8; ++i) { r[i] = lo[i]; r[i + 8] = hi[i]; }
  return r;
}

// Load a 16x32 (f16) WMMA fragment from a row-major [16][pitch] tile.
// Per-lane mapping (wave32): lanes 0-15 -> rows, K-runs {k0..k0+7, k0+16..k0+23};
// lanes 16-31 -> same rows, K-runs {k0+8.., k0+24..}. Two 16B loads per lane.
static __device__ __forceinline__ v16h ldfrag(const _Float16* row, int k0, int sel) {
  v8h lo = *(const v8h*)(row + k0 + sel * 8);
  v8h hi = *(const v8h*)(row + k0 + 16 + sel * 8);
  return mk16(lo, hi);
}

static __device__ __forceinline__ v8f wmma16(v16h a, v16h b, v8f c) {
  return __builtin_amdgcn_wmma_f32_16x16x32_f16(false, a, false, b, (short)0, c,
                                                false, false);
}

static __device__ __forceinline__ float sigm(float x) {
  return 1.0f / (1.0f + __expf(-x));
}

// ---------------------------------------------------------------------------
// One-shot weight / input repack to f16 (grid-stride).
// ---------------------------------------------------------------------------
__global__ void lstm_pack_kernel(const float* __restrict__ x,
                                 const float* __restrict__ Wih0,
                                 const float* __restrict__ Whh0,
                                 const float* __restrict__ bih0,
                                 const float* __restrict__ bhh0,
                                 const float* __restrict__ Wih1,
                                 const float* __restrict__ Whh1,
                                 const float* __restrict__ bih1,
                                 const float* __restrict__ bhh1,
                                 const float* __restrict__ fc1w,
                                 const float* __restrict__ fc2w,
                                 char* __restrict__ ws) {
  _Float16* wcat0 = (_Float16*)(ws + WCAT0_OFF);
  _Float16* wcat1 = (_Float16*)(ws + WCAT1_OFF);
  _Float16* fc1h  = (_Float16*)(ws + FC1W_OFF);
  float*    b0    = (float*)(ws + BIAS0_OFF);
  float*    b1    = (float*)(ws + BIAS1_OFF);
  float*    f2    = (float*)(ws + FC2W_OFF);
  _Float16* xp    = (_Float16*)(ws + XPACK_OFF);

  const long gtid = (long)blockIdx.x * blockDim.x + threadIdx.x;
  const long gs   = (long)gridDim.x * blockDim.x;

  for (long i = gtid; i < 1024L * 288; i += gs) {
    int n = (int)(i / 288), k = (int)(i % 288);
    float v = (k < 256) ? Whh0[n * 256 + k]
                        : ((k < 272) ? Wih0[n * 16 + (k - 256)] : 0.0f);
    wcat0[i] = (_Float16)v;
  }
  for (long i = gtid; i < 1024L * 512; i += gs) {
    int n = (int)(i / 512), k = (int)(i % 512);
    float v = (k < 256) ? Whh1[n * 256 + k] : Wih1[n * 256 + (k - 256)];
    wcat1[i] = (_Float16)v;
  }
  for (long i = gtid; i < 256L * 256; i += gs) fc1h[i] = (_Float16)fc1w[i];
  for (long i = gtid; i < 1024; i += gs) {
    b0[i] = bih0[i] + bhh0[i];
    b1[i] = bih1[i] + bhh1[i];
  }
  for (long i = gtid; i < 256; i += gs) f2[i] = fc2w[i];
  for (long i = gtid; i < 1024L * 256 * 32; i += gs) {
    int t = (int)(i / (256 * 32));
    int r = (int)(i % (256 * 32));
    int b = r >> 5, k = r & 31;
    xp[i] = (k < 16) ? (_Float16)x[((long)b * Tn + t) * 16 + k] : (_Float16)0.0f;
  }
}

// ---------------------------------------------------------------------------
// Fused persistent 2-layer LSTM + fc1/ReLU/fc2 head.
// grid = 16 workgroups (one 16-row batch tile each), block = 512 (16 waves).
// ---------------------------------------------------------------------------
__global__ __launch_bounds__(512) void lstm_fused_kernel(
    const char* __restrict__ ws, const float* __restrict__ fc1b_g,
    const float* __restrict__ fc2b_g, float* __restrict__ out) {
  const _Float16* WCAT0 = (const _Float16*)(ws + WCAT0_OFF);
  const _Float16* WCAT1 = (const _Float16*)(ws + WCAT1_OFF);
  const _Float16* FC1W  = (const _Float16*)(ws + FC1W_OFF);
  const float*    BIAS0 = (const float*)(ws + BIAS0_OFF);
  const float*    BIAS1 = (const float*)(ws + BIAS1_OFF);
  const float*    FC2W  = (const float*)(ws + FC2W_OFF);
  const _Float16* XPACK = (const _Float16*)(ws + XPACK_OFF);

  __shared__ _Float16 lds0[16 * P0];  // layer0 A: [16][0..255]=h0, [256..287]=x_t
  __shared__ _Float16 lds1[16 * P1];  // layer1 A: [16][0..255]=h1, [256..511]=h0_t
  __shared__ float    yacc[16];

  const int tid  = threadIdx.x;
  const int wv   = tid >> 5;
  const int lane = tid & 31;
  const int lh   = lane & 15;
  const int sel  = lane >> 4;
  const int b0r  = blockIdx.x * 16;   // batch tile base
  const int nloc = wv * 16 + lh;      // this lane's column within a 256-wide slice

  for (int i = tid; i < 16 * P0; i += 512) lds0[i] = (_Float16)0.0f;
  for (int i = tid; i < 16 * P1; i += 512) lds1[i] = (_Float16)0.0f;
  __syncthreads();

  // per-lane loop-invariant scalars
  const float bi0 = BIAS0[nloc],       bf0 = BIAS0[256 + nloc];
  const float bg0 = BIAS0[512 + nloc], bo0 = BIAS0[768 + nloc];
  const float bi1 = BIAS1[nloc],       bf1 = BIAS1[256 + nloc];
  const float bg1 = BIAS1[512 + nloc], bo1 = BIAS1[768 + nloc];
  const float fb1 = fc1b_g[nloc];
  const float w2  = FC2W[nloc];
  const float b2  = fc2b_g[0];

  // loop-invariant weight-row pointers (B fragments, [N][K] row-major)
  const _Float16* w0i = WCAT0 + (size_t)(nloc)*288;
  const _Float16* w0f = WCAT0 + (size_t)(256 + nloc) * 288;
  const _Float16* w0g = WCAT0 + (size_t)(512 + nloc) * 288;
  const _Float16* w0o = WCAT0 + (size_t)(768 + nloc) * 288;
  const _Float16* w1i = WCAT1 + (size_t)(nloc)*512;
  const _Float16* w1f = WCAT1 + (size_t)(256 + nloc) * 512;
  const _Float16* w1g = WCAT1 + (size_t)(512 + nloc) * 512;
  const _Float16* w1o = WCAT1 + (size_t)(768 + nloc) * 512;
  const _Float16* wfc = FC1W + (size_t)nloc * 256;
  const _Float16* a0  = lds0 + lh * P0;
  const _Float16* a1  = lds1 + lh * P1;

  float c0r[8], c1r[8];
#pragma unroll
  for (int r = 0; r < 8; ++r) { c0r[r] = 0.0f; c1r[r] = 0.0f; }

  // warm the first x-tile into cache before the recurrence starts
  if (tid < 16)
    __builtin_prefetch(XPACK + ((size_t)0 * Bsz + b0r + tid) * 32, 0, 3);

  for (int t = 0; t < Tn; ++t) {
    // --- stage x_t tile into lds0[:,256:288) (32 halves/row, incl. zero pad)
    if (tid < 256) {
      int row = tid >> 4, q = tid & 15;
      const unsigned* src =
          (const unsigned*)(XPACK + ((size_t)t * Bsz + b0r + row) * 32);
      unsigned* dst = (unsigned*)(lds0 + row * P0 + 256);
      dst[q] = src[q];
    }
    // prefetch next step's tile; an entire step of WMMAs covers the latency
    if (tid < 16 && t + 1 < Tn)
      __builtin_prefetch(XPACK + ((size_t)(t + 1) * Bsz + b0r + tid) * 32, 0, 3);
    __syncthreads();

    // --- layer 0 GEMM: gates = [h0 | x_t] @ [Whh0 | Wih0]^T + bias   (K=288)
    v8f ai, af, ag, ao;
#pragma unroll
    for (int r = 0; r < 8; ++r) { ai[r] = bi0; af[r] = bf0; ag[r] = bg0; ao[r] = bo0; }
#pragma unroll 3
    for (int kk = 0; kk < 9; ++kk) {
      const int k0 = kk * 32;
      v16h A = ldfrag(a0, k0, sel);
      ai = wmma16(A, ldfrag(w0i, k0, sel), ai);
      af = wmma16(A, ldfrag(w0f, k0, sel), af);
      ag = wmma16(A, ldfrag(w0g, k0, sel), ag);
      ao = wmma16(A, ldfrag(w0o, k0, sel), ao);
    }
    _Float16 h0v[8];
#pragma unroll
    for (int r = 0; r < 8; ++r) {
      float iv = sigm(ai[r]), fv = sigm(af[r]);
      float gv = tanhf(ag[r]), ov = sigm(ao[r]);
      float c  = fv * c0r[r] + iv * gv;
      c0r[r]   = c;
      h0v[r]   = (_Float16)(ov * tanhf(c));
    }
    __syncthreads();  // all layer0 A-tile reads complete
#pragma unroll
    for (int r = 0; r < 8; ++r) {
      const int m = r + sel * 8;                 // C/D row mapping
      lds0[m * P0 + nloc]       = h0v[r];        // next step's layer0 h
      lds1[m * P1 + 256 + nloc] = h0v[r];        // layer1 input this step
    }
    __syncthreads();

    // --- layer 1 GEMM: gates = [h1 | h0_t] @ [Whh1 | Wih1]^T + bias  (K=512)
#pragma unroll
    for (int r = 0; r < 8; ++r) { ai[r] = bi1; af[r] = bf1; ag[r] = bg1; ao[r] = bo1; }
#pragma unroll 4
    for (int kk = 0; kk < 16; ++kk) {
      const int k0 = kk * 32;
      v16h A = ldfrag(a1, k0, sel);
      ai = wmma16(A, ldfrag(w1i, k0, sel), ai);
      af = wmma16(A, ldfrag(w1f, k0, sel), af);
      ag = wmma16(A, ldfrag(w1g, k0, sel), ag);
      ao = wmma16(A, ldfrag(w1o, k0, sel), ao);
    }
    _Float16 h1v[8];
#pragma unroll
    for (int r = 0; r < 8; ++r) {
      float iv = sigm(ai[r]), fv = sigm(af[r]);
      float gv = tanhf(ag[r]), ov = sigm(ao[r]);
      float c  = fv * c1r[r] + iv * gv;
      c1r[r]   = c;
      h1v[r]   = (_Float16)(ov * tanhf(c));
    }
    __syncthreads();  // all layer1 A-tile reads complete
#pragma unroll
    for (int r = 0; r < 8; ++r) lds1[(r + sel * 8) * P1 + nloc] = h1v[r];
    if (tid < 16) yacc[tid] = 0.0f;
    __syncthreads();

    // --- fused head: y = fc2 . relu(fc1 @ h1_t + fc1_b) + fc2_b      (K=256)
    v8f z;
#pragma unroll
    for (int r = 0; r < 8; ++r) z[r] = fb1;
#pragma unroll 2
    for (int kk = 0; kk < 8; ++kk) {
      const int k0 = kk * 32;
      v16h A = ldfrag(a1, k0, sel);
      z = wmma16(A, ldfrag(wfc, k0, sel), z);
    }
    float p[8];
#pragma unroll
    for (int r = 0; r < 8; ++r) p[r] = fmaxf(z[r], 0.0f) * w2;
#pragma unroll
    for (int r = 0; r < 8; ++r) {  // reduce over the wave's 16 N-columns
      p[r] += __shfl_xor(p[r], 1, 32);
      p[r] += __shfl_xor(p[r], 2, 32);
      p[r] += __shfl_xor(p[r], 4, 32);
      p[r] += __shfl_xor(p[r], 8, 32);
    }
    if (lh == 0) {
#pragma unroll
      for (int r = 0; r < 8; ++r) atomicAdd(&yacc[r + sel * 8], p[r]);
    }
    __syncthreads();
    if (tid < 16) out[(size_t)(b0r + tid) * Tn + t] = yacc[tid] + b2;
    __syncthreads();
  }
}

extern "C" void kernel_launch(void* const* d_in, const int* in_sizes, int n_in,
                              void* d_out, int out_size, void* d_ws,
                              size_t ws_size, hipStream_t stream) {
  (void)in_sizes; (void)n_in; (void)out_size; (void)ws_size;
  const float* x     = (const float*)d_in[0];
  const float* Wih0  = (const float*)d_in[1];
  const float* Whh0  = (const float*)d_in[2];
  const float* bih0  = (const float*)d_in[3];
  const float* bhh0  = (const float*)d_in[4];
  const float* Wih1  = (const float*)d_in[5];
  const float* Whh1  = (const float*)d_in[6];
  const float* bih1  = (const float*)d_in[7];
  const float* bhh1  = (const float*)d_in[8];
  const float* fc1w  = (const float*)d_in[9];
  const float* fc1b  = (const float*)d_in[10];
  const float* fc2w  = (const float*)d_in[11];
  const float* fc2b  = (const float*)d_in[12];
  char* ws = (char*)d_ws;

  lstm_pack_kernel<<<1024, 256, 0, stream>>>(x, Wih0, Whh0, bih0, bhh0, Wih1,
                                             Whh1, bih1, bhh1, fc1w, fc2w, ws);
  lstm_fused_kernel<<<16, 512, 0, stream>>>(ws, fc1b, fc2b, (float*)d_out);
}
